// Transformer_89575837925826
// MI455X (gfx1250) — matrix operand order repaired and checked by reading, not compiled, not run on hardware
//
#include <hip/hip_runtime.h>

// ---------------------------------------------------------------------------
// Transformer decoder forward for MI455X (gfx1250, wave32, WMMA bf16).
// GEMM uses GLOBAL_LOAD_ASYNC_TO_LDS_B128 double-buffering (ASYNCcnt), so
// tiles never round-trip through VGPRs; all WMMA operands come from
// two 16B/lane ds_load_b128 reads (k = 8h..8h+7 and 16+8h..16+8h+7).
// ---------------------------------------------------------------------------

#define B_SZ   4
#define SEQ    1024
#define DMODEL 512
#define NHEAD  8
#define DKH    64
#define DFF_   2048
#define NLAYER 4
#define MROWS  (B_SZ * SEQ)          // 4096
#define MASKVAL -2147483648.0f
#define LN_EPS  1e-5f

typedef __bf16 bf16_t;
typedef __attribute__((ext_vector_type(16))) __bf16 bf16x16;
typedef __attribute__((ext_vector_type(8)))  float  f32x8;

// ---------------------------------------------------------------------------
// CDNA5 async memory -> LDS copy (ISA 15.18.3, tracked with ASYNCcnt).
// Low 32 bits of a generic pointer into LDS are the DS address (ISA 10.2).
// ---------------------------------------------------------------------------
__device__ __forceinline__ void async_load16(const void* gaddr, void* lds) {
  const unsigned l = (unsigned)(size_t)lds;
  asm volatile("global_load_async_to_lds_b128 %0, %1, off"
               :: "v"(l), "v"(gaddr) : "memory");
}
__device__ __forceinline__ void wait_async0() {
  asm volatile("s_wait_asynccnt 0x0" ::: "memory");
}

// ---------------------------------------------------------------------------
// Fragment helpers (layouts per CDNA5 ISA 7.12.2, wave32)
// ---------------------------------------------------------------------------

// A operand (16x32 bf16) from row-major [m][k] with leading dim ld (bf16 units).
__device__ __forceinline__ bf16x16 frag_a(const bf16_t* base, int ld) {
  const int lane = threadIdx.x & 31;
  const bf16_t* p = base + (lane & 15) * ld + ((lane >> 4) << 3);
  bf16x16 a;
  *reinterpret_cast<uint4*>(&a)       = *reinterpret_cast<const uint4*>(p);
  *(reinterpret_cast<uint4*>(&a) + 1) = *reinterpret_cast<const uint4*>(p + 16);
  return a;
}

// B operand (32x16) from TRANSPOSED storage: source row-major [n][k] (= B^T).
__device__ __forceinline__ bf16x16 frag_bt(const bf16_t* base, int ld) {
  const int lane = threadIdx.x & 31;
  const bf16_t* p = base + (lane & 15) * ld + ((lane >> 4) << 3);
  bf16x16 b;
  *reinterpret_cast<uint4*>(&b)       = *reinterpret_cast<const uint4*>(p);
  *(reinterpret_cast<uint4*>(&b) + 1) = *reinterpret_cast<const uint4*>(p + 16);
  return b;
}

__device__ __forceinline__ f32x8 wmma_bf16(bf16x16 a, bf16x16 b, f32x8 c) {
  return __builtin_amdgcn_wmma_f32_16x16x32_bf16(false, a, false, b, (short)0, c,
                                                 false, false);
}

// ---------------------------------------------------------------------------
// GEMM: C[M,N] = A[M,K](bf16) * B[K,N] + bias, with B given TRANSPOSED:
// Bt[N][K] row-major (ld = K). Tile 128x128x64, double-buffered LDS filled by
// async-to-LDS DMA. 8 waves, wave = 64x32 patch.
// mode 0: store bf16 ; mode 1: relu -> bf16 ; mode 2: += res(f32) -> f32
// ---------------------------------------------------------------------------
__global__ __launch_bounds__(256, 1) void gemm_bf16_k(
    const bf16_t* __restrict__ A, const bf16_t* __restrict__ Bt,
    const float* __restrict__ bias, const float* __restrict__ res,
    bf16_t* __restrict__ Cb, float* __restrict__ Cf,
    int M, int N, int K, int mode) {
  // smem[0] = A tiles (2 buffers), smem[1] = Bt tiles (2 buffers); the whole
  // 80KB block is reused as the epilogue staging tile.
  __shared__ __align__(16) bf16_t smem[2][2][128][80];
  const int tid  = threadIdx.x;
  const int lane = tid & 31;
  const int wave = tid >> 5;
  const int wm = wave >> 2;         // 0..1 : 64 rows each
  const int wn = wave & 3;          // 0..3 : 32 cols each
  const int ln_ = lane & 15, hi = lane >> 4;
  const int bm = blockIdx.y, bn = blockIdx.x;
  (void)M;

  const f32x8 zero8 = {0.f,0.f,0.f,0.f,0.f,0.f,0.f,0.f};
  f32x8 acc[4][2];
#pragma unroll
  for (int mi = 0; mi < 4; ++mi)
#pragma unroll
    for (int ni = 0; ni < 2; ++ni) acc[mi][ni] = zero8;

  const int lr  = tid >> 3;         // tile row handled by this thread (+32/i)
  const int lc8 = tid & 7;          // 8-elem column chunk
  const size_t ga = (size_t)(bm * 128) * K + lc8 * 8;
  const size_t gb = (size_t)(bn * 128) * K + lc8 * 8;

  // prologue: DMA stage 0 into LDS buffer 0
#pragma unroll
  for (int i = 0; i < 4; ++i) {
    async_load16(&A [ga + (size_t)(lr + i * 32) * K], &smem[0][0][lr + i * 32][lc8 * 8]);
    async_load16(&Bt[gb + (size_t)(lr + i * 32) * K], &smem[1][0][lr + i * 32][lc8 * 8]);
  }
  wait_async0();
  __syncthreads();

  int buf = 0;
  for (int k0 = 0; k0 < K; k0 += 64) {
    if (k0 + 64 < K) {              // DMA next stage into the other buffer
#pragma unroll
      for (int i = 0; i < 4; ++i) {
        async_load16(&A [ga + (size_t)(lr + i * 32) * K + k0 + 64],
                     &smem[0][buf ^ 1][lr + i * 32][lc8 * 8]);
        async_load16(&Bt[gb + (size_t)(lr + i * 32) * K + k0 + 64],
                     &smem[1][buf ^ 1][lr + i * 32][lc8 * 8]);
      }
    }
    // compute current stage from LDS (hides the async DMA latency)
#pragma unroll
    for (int ks = 0; ks < 2; ++ks) {
      bf16x16 af[4];
#pragma unroll
      for (int mi = 0; mi < 4; ++mi)
        af[mi] = frag_a(&smem[0][buf][wm * 64 + mi * 16][ks * 32], 80);
#pragma unroll
      for (int ni = 0; ni < 2; ++ni) {
        bf16x16 bfr = frag_bt(&smem[1][buf][wn * 32 + ni * 16][ks * 32], 80);
#pragma unroll
        for (int mi = 0; mi < 4; ++mi)
          acc[mi][ni] = wmma_bf16(af[mi], bfr, acc[mi][ni]);
      }
    }
    wait_async0();
    __syncthreads();
    buf ^= 1;
  }

  // ---------------- epilogue: bounce C tile through LDS ----------------
  if (mode == 2) {
    float* sCf = reinterpret_cast<float*>(&smem[0][0][0][0]);  // [128][136] f32
#pragma unroll
    for (int ni = 0; ni < 2; ++ni) {
      const int col = wn * 32 + ni * 16 + ln_;
      const float bv = bias ? bias[bn * 128 + col] : 0.f;
#pragma unroll
      for (int mi = 0; mi < 4; ++mi)
#pragma unroll
        for (int r = 0; r < 8; ++r) {
          const int row = wm * 64 + mi * 16 + r + hi * 8;
          sCf[row * 136 + col] = acc[mi][ni][r] + bv;
        }
    }
    __syncthreads();
#pragma unroll
    for (int i = 0; i < 16; ++i) {              // 4096 float4 slots
      const int v = i * 256 + tid, r = v >> 5, c4 = v & 31;
      const size_t g = (size_t)(bm * 128 + r) * N + bn * 128 + c4 * 4;
      float4 rv = *(const float4*)&res[g];
      const float4 av = *(const float4*)&sCf[r * 136 + c4 * 4];
      rv.x += av.x; rv.y += av.y; rv.z += av.z; rv.w += av.w;
      *(float4*)&Cf[g] = rv;
    }
  } else {
    bf16_t* sC = &smem[0][0][0][0];             // [128][136] bf16
#pragma unroll
    for (int ni = 0; ni < 2; ++ni) {
      const int col = wn * 32 + ni * 16 + ln_;
      const float bv = bias ? bias[bn * 128 + col] : 0.f;
#pragma unroll
      for (int mi = 0; mi < 4; ++mi)
#pragma unroll
        for (int r = 0; r < 8; ++r) {
          const int row = wm * 64 + mi * 16 + r + hi * 8;
          float v = acc[mi][ni][r] + bv;
          if (mode == 1) v = fmaxf(v, 0.f);
          sC[row * 136 + col] = (bf16_t)v;
        }
    }
    __syncthreads();
#pragma unroll
    for (int i = 0; i < 8; ++i) {               // 2048 8-elem vectors
      const int v = i * 256 + tid, r = v >> 4, c8 = v & 15;
      *(uint4*)&Cb[(size_t)(bm * 128 + r) * N + bn * 128 + c8 * 8] =
          *(const uint4*)&sC[r * 136 + c8 * 8];
    }
  }
}

// ---------------------------------------------------------------------------
// Flash attention: one block = (b, h, 64-row q tile); 8 waves; DK = 64.
// ---------------------------------------------------------------------------
__global__ __launch_bounds__(256, 1) void attention_k(
    const bf16_t* __restrict__ Qg, int ldq, const bf16_t* __restrict__ Kg, int ldk,
    const bf16_t* __restrict__ Vg, int ldv, bf16_t* __restrict__ Og, int ldo,
    int causal) {
  __shared__ bf16_t qt[64][80];
  __shared__ bf16_t kt[64][80];
  __shared__ bf16_t vt[64][80];    // transposed: vt[dk][key]
  __shared__ bf16_t pt[64][80];    // probabilities (bf16 A operand for P@V)
  __shared__ float  sc[64][68];    // raw scores
  __shared__ float  mrow[64], lrow[64], srow[64];
  __shared__ float  pmax[4][64], psum[4][64];

  const int tid  = threadIdx.x;
  const int lane = tid & 31;
  const int wave = tid >> 5;
  const int wm = wave >> 1;        // 0..3 : 16 rows each
  const int wn = wave & 1;         // 0..1 : 32 cols each
  const int ln_ = lane & 15, hi = lane >> 4;
  const int r4 = tid & 63, q4 = tid >> 6;   // softmax: 4 threads per row
  const int qt_i = blockIdx.x, h = blockIdx.y, b = blockIdx.z;
  const int q0 = qt_i * 64;
  const size_t rowbase = (size_t)b * SEQ;
  const int hoff = h * DKH;

  const int lr  = tid >> 3;        // 0..31  (row within half-tile; +32 for hi half)
  const int lc8 = tid & 7;

  // Q tile 64x64 via async DMA (waited before first use below)
  async_load16(&Qg[(rowbase + q0 + lr)      * ldq + hoff + lc8 * 8], &qt[lr][lc8 * 8]);
  async_load16(&Qg[(rowbase + q0 + lr + 32) * ldq + hoff + lc8 * 8], &qt[lr + 32][lc8 * 8]);
  if (tid < 64) { mrow[tid] = -3.4e38f; lrow[tid] = 0.f; }

  const f32x8 zero8 = {0.f,0.f,0.f,0.f,0.f,0.f,0.f,0.f};
  f32x8 oacc[2]; oacc[0] = zero8; oacc[1] = zero8;

  const int ntiles = causal ? (qt_i + 1) : (SEQ / 64);
  for (int t = 0; t < ntiles; ++t) {
    const size_t kbase = (rowbase + t * 64);
    // K tile via async DMA; V tile staged through regs (needs transpose)
    async_load16(&Kg[(kbase + lr)      * ldk + hoff + lc8 * 8], &kt[lr][lc8 * 8]);
    async_load16(&Kg[(kbase + lr + 32) * ldk + hoff + lc8 * 8], &kt[lr + 32][lc8 * 8]);
    uint4 v0v = *(const uint4*)&Vg[(kbase + lr)      * ldv + hoff + lc8 * 8];
    uint4 v1v = *(const uint4*)&Vg[(kbase + lr + 32) * ldv + hoff + lc8 * 8];
    if (t + 1 < ntiles) {                         // prefetch next tile
      __builtin_prefetch(&Kg[(kbase + 64 + lr) * ldk + hoff + lc8 * 8], 0, 1);
      __builtin_prefetch(&Vg[(kbase + 64 + lr) * ldv + hoff + lc8 * 8], 0, 1);
    }
    {                                             // scatter V transposed
      const bf16_t* p0 = (const bf16_t*)&v0v;
      const bf16_t* p1 = (const bf16_t*)&v1v;
#pragma unroll
      for (int j = 0; j < 8; ++j) {
        vt[lc8 * 8 + j][lr]      = p0[j];
        vt[lc8 * 8 + j][lr + 32] = p1[j];
      }
    }
    wait_async0();
    __syncthreads();

    // S = Q K^T / 8
    f32x8 sacc[2]; sacc[0] = zero8; sacc[1] = zero8;
#pragma unroll
    for (int ks = 0; ks < 2; ++ks) {
      bf16x16 a = frag_a(&qt[wm * 16][ks * 32], 80);
#pragma unroll
      for (int ni = 0; ni < 2; ++ni) {
        bf16x16 bb = frag_bt(&kt[wn * 32 + ni * 16][ks * 32], 80);
        sacc[ni] = wmma_bf16(a, bb, sacc[ni]);
      }
    }
#pragma unroll
    for (int ni = 0; ni < 2; ++ni)
#pragma unroll
      for (int r = 0; r < 8; ++r) {
        const int row = wm * 16 + r + hi * 8;
        const int col = wn * 32 + ni * 16 + ln_;
        float v = sacc[ni][r] * 0.125f;
        if (causal && (t * 64 + col) > (q0 + row)) v += MASKVAL;
        sc[row][col] = v;
      }
    __syncthreads();

    // streaming softmax, 4 threads per row
    {
      float mx = -3.4e38f;
#pragma unroll
      for (int c = 0; c < 16; ++c) mx = fmaxf(mx, sc[r4][q4 * 16 + c]);
      pmax[q4][r4] = mx;
    }
    __syncthreads();
    if (q4 == 0) {
      const float mo = mrow[r4];
      float mx = fmaxf(fmaxf(pmax[0][r4], pmax[1][r4]),
                       fmaxf(pmax[2][r4], pmax[3][r4]));
      mx = fmaxf(mo, mx);
      srow[r4] = __expf(mo - mx);
      mrow[r4] = mx;
    }
    __syncthreads();
    {
      const float mx = mrow[r4];
      float s = 0.f;
#pragma unroll
      for (int c = 0; c < 16; ++c) {
        const float p = __expf(sc[r4][q4 * 16 + c] - mx);
        pt[r4][q4 * 16 + c] = (bf16_t)p;
        s += p;
      }
      psum[q4][r4] = s;
    }
    __syncthreads();
    if (q4 == 0)
      lrow[r4] = lrow[r4] * srow[r4] +
                 (psum[0][r4] + psum[1][r4]) + (psum[2][r4] + psum[3][r4]);

    // rescale O, then O += P V
    float al[8];
#pragma unroll
    for (int r = 0; r < 8; ++r) al[r] = srow[wm * 16 + r + hi * 8];
#pragma unroll
    for (int ni = 0; ni < 2; ++ni)
#pragma unroll
      for (int r = 0; r < 8; ++r) oacc[ni][r] *= al[r];

#pragma unroll
    for (int ks = 0; ks < 2; ++ks) {
      bf16x16 a = frag_a(&pt[wm * 16][ks * 32], 80);
#pragma unroll
      for (int ni = 0; ni < 2; ++ni) {
        bf16x16 bb = frag_bt(&vt[wn * 32 + ni * 16][ks * 32], 80);
        oacc[ni] = wmma_bf16(a, bb, oacc[ni]);
      }
    }
    __syncthreads();
  }

#pragma unroll
  for (int ni = 0; ni < 2; ++ni)
#pragma unroll
    for (int r = 0; r < 8; ++r) {
      const int row = wm * 16 + r + hi * 8;
      const int col = wn * 32 + ni * 16 + ln_;
      const float v = oacc[ni][r] / lrow[row];
      Og[(rowbase + q0 + row) * ldo + hoff + col] = (bf16_t)v;
    }
}

// ---------------------------------------------------------------------------
// Embedding + sinusoidal PE:  v = x[b,s]*w[d] + bias[d] + PE[s,d]
// ---------------------------------------------------------------------------
__global__ void embed_k(const float* __restrict__ x, const float* __restrict__ w,
                        const float* __restrict__ bias, bf16_t* __restrict__ ob,
                        float* __restrict__ of, int n) {
  const int idx = blockIdx.x * blockDim.x + threadIdx.x;
  if (idx >= n) return;
  const int d = idx & (DMODEL - 1);
  const int s = (idx >> 9) & (SEQ - 1);
  const int b = idx >> 19;
  const float theta = (float)s * __powf(10000.0f, -(float)(d & ~1) * (1.0f / DMODEL));
  const float pe = (d & 1) ? __cosf(theta) : __sinf(theta);
  const float v = x[b * SEQ + s] * w[d] + bias[d] + pe;
  ob[idx] = (bf16_t)v;
  if (of) of[idx] = v;
}

// ---------------------------------------------------------------------------
// LayerNorm over D=512, one 128-thread block per row; writes f32 + bf16.
// ---------------------------------------------------------------------------
__global__ __launch_bounds__(128) void layernorm_k(
    const float* __restrict__ x, const float* __restrict__ g,
    const float* __restrict__ bta, float* __restrict__ yf, bf16_t* __restrict__ yb) {
  __shared__ float red[128];
  const int row = blockIdx.x, t = threadIdx.x;
  const float* xr = x + (size_t)row * DMODEL;
  float v[4], s = 0.f;
#pragma unroll
  for (int j = 0; j < 4; ++j) { v[j] = xr[t + 128 * j]; s += v[j]; }
  red[t] = s; __syncthreads();
  for (int o = 64; o > 0; o >>= 1) { if (t < o) red[t] += red[t + o]; __syncthreads(); }
  const float mu = red[0] * (1.0f / DMODEL);
  __syncthreads();
  s = 0.f;
#pragma unroll
  for (int j = 0; j < 4; ++j) { float d = v[j] - mu; s += d * d; }
  red[t] = s; __syncthreads();
  for (int o = 64; o > 0; o >>= 1) { if (t < o) red[t] += red[t + o]; __syncthreads(); }
  const float rstd = rsqrtf(red[0] * (1.0f / DMODEL) + LN_EPS);
#pragma unroll
  for (int j = 0; j < 4; ++j) {
    const int d = t + 128 * j;
    const float y = (v[j] - mu) * rstd * g[d] + bta[d];
    yf[(size_t)row * DMODEL + d] = y;
    yb[(size_t)row * DMODEL + d] = (bf16_t)y;
  }
}

// ---------------------------------------------------------------------------
// Weight repack: fp32 W[K][N] (ld sld) -> bf16 W^T[N][K] (coalesced dst writes)
// ---------------------------------------------------------------------------
__global__ void convertT_k(const float* __restrict__ src, bf16_t* __restrict__ dst,
                           int K_, int N_, int sld) {
  const int idx = blockIdx.x * blockDim.x + threadIdx.x;
  if (idx >= K_ * N_) return;
  const int n = idx / K_, k = idx - n * K_;
  dst[idx] = (bf16_t)src[(size_t)k * sld + n];
}

__global__ void copy_f32_k(const float* __restrict__ src, float* __restrict__ dst, int n) {
  const int i = blockIdx.x * blockDim.x + threadIdx.x;
  if (i < n) dst[i] = src[i];
}

// AvgPool(dec @ fin_w + fin_b) == dec . rowmean(fin_w) + mean(fin_b)
__global__ void prep_final_k(const float* __restrict__ fw, const float* __restrict__ fb,
                             float* __restrict__ favg) {
  const int k = blockIdx.x * blockDim.x + threadIdx.x;
  if (k < DMODEL) {
    float s = 0.f;
    for (int j = 0; j < DMODEL; ++j) s += fw[(size_t)k * DMODEL + j];
    favg[k] = s * (1.0f / DMODEL);
  }
  if (k == 0) {
    float s = 0.f;
    for (int j = 0; j < DMODEL; ++j) s += fb[j];
    favg[DMODEL] = s * (1.0f / DMODEL);
  }
}

__global__ __launch_bounds__(128) void final_out_k(
    const float* __restrict__ dec, const float* __restrict__ favg,
    float* __restrict__ out) {
  __shared__ float red[128];
  const int row = blockIdx.x, t = threadIdx.x;
  const float* xr = dec + (size_t)row * DMODEL;
  float s = 0.f;
#pragma unroll
  for (int j = 0; j < 4; ++j) { const int d = t + 128 * j; s += xr[d] * favg[d]; }
  red[t] = s; __syncthreads();
  for (int o = 64; o > 0; o >>= 1) { if (t < o) red[t] += red[t + o]; __syncthreads(); }
  if (t == 0) out[row] = red[0] + favg[DMODEL];
}

// ---------------------------------------------------------------------------
// Host orchestration
// ---------------------------------------------------------------------------
extern "C" void kernel_launch(void* const* d_in, const int* in_sizes, int n_in,
                              void* d_out, int out_size, void* d_ws, size_t ws_size,
                              hipStream_t stream) {
  (void)in_sizes; (void)n_in; (void)out_size; (void)ws_size;
  const float* in_enc    = (const float*)d_in[0];
  const float* in_dec    = (const float*)d_in[1];
  const float* emb_in_w  = (const float*)d_in[2];
  const float* emb_in_b  = (const float*)d_in[3];
  const float* emb_out_w = (const float*)d_in[4];
  const float* emb_out_b = (const float*)d_in[5];
  const float* sa_wq = (const float*)d_in[6];   const float* sa_bq = (const float*)d_in[7];
  const float* sa_wk = (const float*)d_in[8];   const float* sa_bk = (const float*)d_in[9];
  const float* sa_wv = (const float*)d_in[10];  const float* sa_bv = (const float*)d_in[11];
  const float* sa_wo = (const float*)d_in[12];  const float* sa_bo = (const float*)d_in[13];
  const float* ca_wq = (const float*)d_in[14];  const float* ca_bq = (const float*)d_in[15];
  const float* ca_wk = (const float*)d_in[16];  const float* ca_bk = (const float*)d_in[17];
  const float* ca_wv = (const float*)d_in[18];  const float* ca_bv = (const float*)d_in[19];
  const float* ca_wo = (const float*)d_in[20];  const float* ca_bo = (const float*)d_in[21];
  const float* ln_g  = (const float*)d_in[22];  const float* ln_b  = (const float*)d_in[23];
  const float* ff_w1 = (const float*)d_in[24];  const float* ff_b1 = (const float*)d_in[25];
  const float* ff_w2 = (const float*)d_in[26];  const float* ff_b2 = (const float*)d_in[27];
  const float* fin_w = (const float*)d_in[28];  const float* fin_b = (const float*)d_in[29];
  float* out = (float*)d_out;

  char* ws = (char*)d_ws; size_t off = 0;
  auto take = [&](size_t bytes) -> void* {
    void* p = ws + off; off = (off + bytes + 255) & ~(size_t)255; return p;
  };

  bf16_t* enc_b = (bf16_t*)take((size_t)MROWS * DMODEL * 2);
  float*  dec_f = (float*) take((size_t)MROWS * DMODEL * 4);
  bf16_t* dec_b = (bf16_t*)take((size_t)MROWS * DMODEL * 2);
  bf16_t* qkv   = (bf16_t*)take((size_t)MROWS * 3 * DMODEL * 2);
  bf16_t* attb  = (bf16_t*)take((size_t)MROWS * DMODEL * 2);
  bf16_t* qbuf  = (bf16_t*)take((size_t)MROWS * DMODEL * 2);
  bf16_t* kvbuf = (bf16_t*)take((size_t)MROWS * 2 * DMODEL * 2);
  bf16_t* hid   = (bf16_t*)take((size_t)MROWS * DFF_ * 2);
  float*  sumf  = (float*) take((size_t)MROWS * DMODEL * 4);
  float*  x1f   = (float*) take((size_t)MROWS * DMODEL * 4);
  bf16_t* x1b   = (bf16_t*)take((size_t)MROWS * DMODEL * 2);
  float*  x2f   = (float*) take((size_t)MROWS * DMODEL * 4);
  bf16_t* x2b   = (bf16_t*)take((size_t)MROWS * DMODEL * 2);
  float*  favg  = (float*) take((DMODEL + 1) * 4);

  // Transposed bf16 weights: Wt[N][K] row-major, ld = K.
  bf16_t *Wqkv[NLAYER], *Wsao[NLAYER], *Wcaq[NLAYER], *Wkv[NLAYER],
         *Wcao[NLAYER], *Wf1[NLAYER], *Wf2[NLAYER];
  float *bqkv[NLAYER], *bkv[NLAYER];
  for (int i = 0; i < NLAYER; ++i) {
    Wqkv[i] = (bf16_t*)take((size_t)3 * DMODEL * DMODEL * 2);  // 1536 x 512
    Wsao[i] = (bf16_t*)take((size_t)DMODEL * DMODEL * 2);      // 512 x 512
    Wcaq[i] = (bf16_t*)take((size_t)DMODEL * DMODEL * 2);
    Wkv[i]  = (bf16_t*)take((size_t)2 * DMODEL * DMODEL * 2);  // 1024 x 512
    Wcao[i] = (bf16_t*)take((size_t)DMODEL * DMODEL * 2);
    Wf1[i]  = (bf16_t*)take((size_t)DFF_ * DMODEL * 2);        // 2048 x 512
    Wf2[i]  = (bf16_t*)take((size_t)DMODEL * DFF_ * 2);        // 512 x 2048
    bqkv[i] = (float*)take(3 * DMODEL * 4);
    bkv[i]  = (float*)take(2 * DMODEL * 4);
  }

  auto convT = [&](const float* s, bf16_t* d, int K_, int N_, int sld) {
    const int n = K_ * N_;
    convertT_k<<<(n + 255) / 256, 256, 0, stream>>>(s, d, K_, N_, sld);
  };
  auto cpy = [&](const float* s, float* d, int n) {
    copy_f32_k<<<(n + 255) / 256, 256, 0, stream>>>(s, d, n);
  };
  auto gemm = [&](const bf16_t* A, const bf16_t* Bt, const float* bias,
                  const float* res, bf16_t* Cb, float* Cf, int M_, int N_, int K_,
                  int mode) {
    gemm_bf16_k<<<dim3(N_ / 128, M_ / 128), 256, 0, stream>>>(A, Bt, bias, res, Cb,
                                                              Cf, M_, N_, K_, mode);
  };
  auto attn = [&](const bf16_t* Q, int ldq, const bf16_t* K, int ldk,
                  const bf16_t* V, int ldv, bf16_t* O, int ldo, int causal) {
    attention_k<<<dim3(SEQ / 64, NHEAD, B_SZ), 256, 0, stream>>>(Q, ldq, K, ldk, V,
                                                                 ldv, O, ldo, causal);
  };
  auto lnorm = [&](const float* x, int layer, float* yf, bf16_t* yb) {
    layernorm_k<<<MROWS, 128, 0, stream>>>(x, ln_g + (size_t)layer * DMODEL,
                                           ln_b + (size_t)layer * DMODEL, yf, yb);
  };

  // Embeddings + positional encodings (encoder stack is dead code in the ref:
  // cross-attention consumes the positional-encoded encoder embedding directly).
  {
    const int n = MROWS * DMODEL;
    embed_k<<<(n + 255) / 256, 256, 0, stream>>>(in_enc, emb_in_w, emb_in_b,
                                                 enc_b, (float*)nullptr, n);
    embed_k<<<(n + 255) / 256, 256, 0, stream>>>(in_dec, emb_out_w, emb_out_b,
                                                 dec_b, dec_f, n);
  }

  // Repack weights: bf16 + transpose (QKV fused; cross-attn K/V fused).
  const int DD = DMODEL * DMODEL;
  for (int i = 0; i < NLAYER; ++i) {
    const size_t wo = (size_t)i * DD;
    convT(sa_wq + wo, Wqkv[i] + 0 * DD, DMODEL, DMODEL, DMODEL);   // rows 0..511
    convT(sa_wk + wo, Wqkv[i] + 1 * DD, DMODEL, DMODEL, DMODEL);   // rows 512..1023
    convT(sa_wv + wo, Wqkv[i] + 2 * DD, DMODEL, DMODEL, DMODEL);   // rows 1024..1535
    convT(sa_wo + wo, Wsao[i], DMODEL, DMODEL, DMODEL);
    convT(ca_wq + wo, Wcaq[i], DMODEL, DMODEL, DMODEL);
    convT(ca_wk + wo, Wkv[i] + 0 * DD, DMODEL, DMODEL, DMODEL);
    convT(ca_wv + wo, Wkv[i] + 1 * DD, DMODEL, DMODEL, DMODEL);
    convT(ca_wo + wo, Wcao[i], DMODEL, DMODEL, DMODEL);
    convT(ff_w1 + (size_t)i * DMODEL * DFF_, Wf1[i], DMODEL, DFF_, DFF_);
    convT(ff_w2 + (size_t)i * DFF_ * DMODEL, Wf2[i], DFF_, DMODEL, DMODEL);
    cpy(sa_bq + (size_t)i * DMODEL, bqkv[i] + 0,          DMODEL);
    cpy(sa_bk + (size_t)i * DMODEL, bqkv[i] + DMODEL,     DMODEL);
    cpy(sa_bv + (size_t)i * DMODEL, bqkv[i] + 2 * DMODEL, DMODEL);
    cpy(ca_bk + (size_t)i * DMODEL, bkv[i] + 0,      DMODEL);
    cpy(ca_bv + (size_t)i * DMODEL, bkv[i] + DMODEL, DMODEL);
  }
  prep_final_k<<<2, 256, 0, stream>>>(fin_w, fin_b, favg);

  for (int i = 0; i < NLAYER; ++i) {
    // --- masked self attention ---
    gemm(dec_b, Wqkv[i], bqkv[i], nullptr, qkv, nullptr, MROWS, 3 * DMODEL, DMODEL, 0);
    attn(qkv, 3 * DMODEL, qkv + DMODEL, 3 * DMODEL, qkv + 2 * DMODEL, 3 * DMODEL,
         attb, DMODEL, 1);
    gemm(attb, Wsao[i], sa_bo + (size_t)i * DMODEL, dec_f, nullptr, sumf,
         MROWS, DMODEL, DMODEL, 2);
    lnorm(sumf, i, x1f, x1b);
    // --- cross attention (over positional-encoded encoder embedding) ---
    gemm(x1b, Wcaq[i], ca_bq + (size_t)i * DMODEL, nullptr, qbuf, nullptr,
         MROWS, DMODEL, DMODEL, 0);
    gemm(enc_b, Wkv[i], bkv[i], nullptr, kvbuf, nullptr, MROWS, 2 * DMODEL, DMODEL, 0);
    attn(qbuf, DMODEL, kvbuf, 2 * DMODEL, kvbuf + DMODEL, 2 * DMODEL, attb, DMODEL, 0);
    gemm(attb, Wcao[i], ca_bo + (size_t)i * DMODEL, x1f, nullptr, sumf,
         MROWS, DMODEL, DMODEL, 2);
    lnorm(sumf, i, x2f, x2b);
    // --- FFN ---
    gemm(x2b, Wf1[i], ff_b1 + (size_t)i * DFF_, nullptr, hid, nullptr,
         MROWS, DFF_, DMODEL, 1);
    gemm(hid, Wf2[i], ff_b2 + (size_t)i * DMODEL, x2f, nullptr, sumf,
         MROWS, DMODEL, DFF_, 2);
    lnorm(sumf, i, dec_f, dec_b);
  }

  final_out_k<<<MROWS, 128, 0, stream>>>(dec_f, favg, out);
}
